// SMPL_15367392985468
// MI455X (gfx1250) — matrix operand order repaired and checked by reading, not compiled, not run on hardware
//
#include <hip/hip_runtime.h>
#include <hip/hip_bf16.h>

typedef __attribute__((ext_vector_type(16))) _Float16 v16h;
typedef __attribute__((ext_vector_type(8)))  float    v8f;

#define NVV 778
#define NB_ 400
#define NCOL 2334          // NV*3
#define NCOLP 2336         // padded to 146*16
#define NTILES_N 146
#define KS 32              // padded K for shapedirs (10 -> 32)
#define KP 160             // padded K for posedirs (135 -> 160)

// ---------------- workspace layout (bytes, 256-aligned) ----------------
static constexpr size_t OFF_RALL   = 0;                         // 400*16*9 f32   = 230400
static constexpr size_t OFF_PF16   = 230400;                    // 400*160 f16    = 128000
static constexpr size_t OFF_BETA16 = 358400;                    // 400*32  f16    = 25600
static constexpr size_t OFF_S16    = 384000;                    // 2336*32 f16    = 149504
static constexpr size_t OFF_P16    = 533504;                    // 2336*160 f16   = 747520
static constexpr size_t OFF_VSH    = 1281024;                   // 400*2334 f32   (pad 3734528)
static constexpr size_t OFF_VPO    = 5015552;                   // 400*2334 f32
static constexpr size_t OFF_J      = 8750080;                   // 400*16*3 f32   = 76800
static constexpr size_t OFF_A      = 8826880;                   // 400*16*16 f32  = 409600
static constexpr size_t OFF_VERTS  = 9236480;                   // 400*2334 f32
static constexpr size_t WS_TOTAL   = 12971008;

// ---------------- f16 conversion / transpose kernels ----------------
__global__ void conv_S_kernel(const float* __restrict__ shapedirs, _Float16* __restrict__ S16) {
    int idx = blockIdx.x * blockDim.x + threadIdx.x;
    if (idx >= NCOLP * KS) return;
    int n = idx / KS, k = idx % KS;
    float v = (k < 10 && n < NCOL) ? shapedirs[k * NCOL + n] : 0.0f;
    S16[n * KS + k] = (_Float16)v;
}

__global__ void conv_P_kernel(const float* __restrict__ posedirs, _Float16* __restrict__ P16) {
    int idx = blockIdx.x * blockDim.x + threadIdx.x;
    if (idx >= NCOLP * KP) return;
    int n = idx / KP, k = idx % KP;
    float v = (k < 135 && n < NCOL) ? posedirs[k * NCOL + n] : 0.0f;
    P16[n * KP + k] = (_Float16)v;
}

// ---------------- per-batch prep: pose, Rodrigues, quat2mat ----------------
__global__ void prep_kernel(const float* __restrict__ theta, const float* __restrict__ quat,
                            const float* __restrict__ beta,
                            const float* __restrict__ hands_comp, const float* __restrict__ hands_mean,
                            float* __restrict__ R_all, _Float16* __restrict__ pf16,
                            _Float16* __restrict__ beta16) {
    int b = blockIdx.x;
    int t = threadIdx.x;
    __shared__ float pose[48];
    if (t < 45) {
        float acc = hands_mean[t];
        for (int k = 0; k < 45; ++k) acc += theta[b * 45 + k] * hands_comp[k * 45 + t];
        pose[t] = acc;
    }
    __syncthreads();
    if (t < 15) {
        float rx = pose[3*t+0], ry = pose[3*t+1], rz = pose[3*t+2];
        float ang = sqrtf(rx*rx + ry*ry + rz*rz + 1e-8f);
        float inv = 1.0f / ang;
        float x = rx*inv, y = ry*inv, z = rz*inv;
        float s = sinf(ang), c = cosf(ang), C = 1.0f - c;
        float R[9];
        R[0] = 1.0f + C * (-(y*y + z*z));
        R[1] = s * (-z) + C * (x*y);
        R[2] = s * ( y) + C * (x*z);
        R[3] = s * ( z) + C * (x*y);
        R[4] = 1.0f + C * (-(x*x + z*z));
        R[5] = s * (-x) + C * (y*z);
        R[6] = s * (-y) + C * (x*z);
        R[7] = s * ( x) + C * (y*z);
        R[8] = 1.0f + C * (-(x*x + y*y));
        float* dst = R_all + (b * 16 + (t + 1)) * 9;
        _Float16* pf = pf16 + b * KP + 9 * t;
        #pragma unroll
        for (int e = 0; e < 9; ++e) {
            dst[e] = R[e];
            float ident = (e == 0 || e == 4 || e == 8) ? 1.0f : 0.0f;
            pf[e] = (_Float16)(R[e] - ident);
        }
    } else if (t == 15) {
        float qw = quat[b*4+0], qx = quat[b*4+1], qy = quat[b*4+2], qz = quat[b*4+3];
        float inv = 1.0f / sqrtf(qw*qw + qx*qx + qy*qy + qz*qz + 1e-8f);
        qw *= inv; qx *= inv; qy *= inv; qz *= inv;
        float w2 = qw*qw, x2 = qx*qx, y2 = qy*qy, z2 = qz*qz;
        float wx = qw*qx, wy = qw*qy, wz = qw*qz;
        float xy = qx*qy, xz = qx*qz, yz = qy*qz;
        float R[9];
        R[0] = w2 + x2 - y2 - z2;  R[1] = 2.0f*(xy - wz);      R[2] = 2.0f*(wy + xz);
        R[3] = 2.0f*(wz + xy);     R[4] = w2 - x2 + y2 - z2;   R[5] = 2.0f*(yz - wx);
        R[6] = 2.0f*(xz - wy);     R[7] = 2.0f*(wx + yz);      R[8] = w2 - x2 - y2 + z2;
        // root_R = R @ diag(1,-1,-1): negate columns 1 and 2
        R[1] = -R[1]; R[4] = -R[4]; R[7] = -R[7];
        R[2] = -R[2]; R[5] = -R[5]; R[8] = -R[8];
        float* dst = R_all + (b * 16 + 0) * 9;
        #pragma unroll
        for (int e = 0; e < 9; ++e) dst[e] = R[e];
    } else if (t < 48) {
        int k = t - 16;
        beta16[b * KS + k] = (_Float16)((k < 10) ? beta[b * 10 + k] : 0.0f);
    } else {
        int i0 = 135 + (t - 48);
        pf16[b * KP + i0] = (_Float16)0.0f;
        int i1 = i0 + 16;
        if (i1 < KP) pf16[b * KP + i1] = (_Float16)0.0f;
    }
}

// ---------------- WMMA GEMM helpers ----------------
union H16 { uint4 q[2]; v16h h; };

// A operand, 16-bit 16x32 layout: lane<16 -> K {0..7,16..23}; lane>=16 -> K {8..15,24..31}
// Works for both global and LDS (generic) pointers; offsets are 16B aligned.
__device__ inline v16h load_a16(const _Float16* base, int row, int strideH, int kc, bool hi) {
    const unsigned* rowp = reinterpret_cast<const unsigned*>(base + (size_t)row * strideH);
    int d0 = kc * 16 + (hi ? 4 : 0);
    H16 t;
    t.q[0] = *reinterpret_cast<const uint4*>(rowp + d0);
    t.q[1] = *reinterpret_cast<const uint4*>(rowp + d0 + 8);
    return t.h;
}

// B operand, 16-bit 32x16: lane<16 -> col N=lane, K {0..7,16..23}; lane>=16 -> K {8..15,24..31}
// (B matrices stored column-major [n][k], so each lane reads two runs of 8 contiguous halves)
__device__ inline v16h load_b16(const _Float16* base, int col, int strideH, int kc, bool hi) {
    const unsigned* colp = reinterpret_cast<const unsigned*>(base + (size_t)col * strideH);
    int d0 = kc * 16 + (hi ? 4 : 0);
    H16 t;
    t.q[0] = *reinterpret_cast<const uint4*>(colp + d0);
    t.q[1] = *reinterpret_cast<const uint4*>(colp + d0 + 8);
    return t.h;
}

// Async global -> LDS copy of one 16-byte chunk (CDNA5 GLOBAL_LOAD_ASYNC_TO_LDS_B128).
__device__ inline void async_copy_b128(const _Float16* g, const _Float16* lds_ptr) {
    unsigned lds_off = (unsigned)(uintptr_t)lds_ptr;
    asm volatile("global_load_async_to_lds_b128 %0, %1, off"
                 :: "v"(lds_off), "v"(g) : "memory");
}

// 8 waves per block: one M-tile (16 batch rows), 8 consecutive N-tiles.
// A tile (beta rows + pose_feature rows) staged in LDS once via async copy,
// shared by all waves; each wave streams its own B columns from global.
__global__ __launch_bounds__(256) void gemm_kernel(const _Float16* __restrict__ beta16,
                                                   const _Float16* __restrict__ pf16,
                                                   const _Float16* __restrict__ S16,
                                                   const _Float16* __restrict__ P16,
                                                   const float* __restrict__ vtempl,
                                                   float* __restrict__ vshaped,
                                                   float* __restrict__ vposed) {
    __shared__ __align__(16) _Float16 sBeta[16 * KS];   // 16 rows x 32 halves  (1 KB)
    __shared__ __align__(16) _Float16 sPf[16 * KP];     // 16 rows x 160 halves (5 KB)

    const int tid   = threadIdx.x;
    const int Mbase = blockIdx.y * 16;

    // ---- async-stage the A tile into LDS: 64 beta chunks + 320 pf chunks of 16B ----
    for (int i = tid; i < 384; i += 256) {
        if (i < 64) {
            int row = i >> 2, c = i & 3;                // 4 chunks per 64B beta row
            async_copy_b128(beta16 + (size_t)(Mbase + row) * KS + c * 8,
                            sBeta + row * KS + c * 8);
        } else {
            int j = i - 64;
            int row = j / 20, c = j % 20;               // 20 chunks per 320B pf row
            async_copy_b128(pf16 + (size_t)(Mbase + row) * KP + c * 8,
                            sPf + row * KP + c * 8);
        }
    }
    asm volatile("s_wait_asynccnt 0" ::: "memory");
    __syncthreads();

    const int wave  = tid >> 5;
    const int lane  = tid & 31;
    const bool hi   = lane >= 16;
    const int l15   = lane & 15;
    const int ntile = blockIdx.x * 8 + wave;
    if (ntile >= NTILES_N) return;                      // whole-wave exit: EXEC stays all-ones
    const int Nbase = ntile * 16;
    const int n = Nbase + l15;                          // padded B arrays go to 2336: loads safe

    __builtin_prefetch(P16 + (size_t)n * KP, 0, 0);

    // ---- v_shaped: one K=32 WMMA (real K=10, rest zero) ----
    v16h a0 = load_a16(sBeta, l15, KS, 0, hi);          // A from LDS (ds_load_b128)
    v16h b0 = load_b16(S16, n, KS, 0, hi);
    v8f c = {};
    c = __builtin_amdgcn_wmma_f32_16x16x32_f16(false, a0, false, b0, (short)0, c, false, false);

    float vt = (n < NCOL) ? vtempl[n] : 0.0f;
    #pragma unroll
    for (int r = 0; r < 8; ++r) c[r] += vt;

    if (n < NCOL) {
        #pragma unroll
        for (int r = 0; r < 8; ++r) {
            int mm = Mbase + r + (hi ? 8 : 0);
            vshaped[(size_t)mm * NCOL + n] = c[r];
        }
    }

    // ---- v_posed: accumulate pf @ P on top of v_shaped (5 x K=32 WMMAs) ----
    #pragma unroll
    for (int kc = 0; kc < 5; ++kc) {
        v16h ap = load_a16(sPf, l15, KP, kc, hi);       // A from LDS
        v16h bp = load_b16(P16, n, KP, kc, hi);
        c = __builtin_amdgcn_wmma_f32_16x16x32_f16(false, ap, false, bp, (short)0, c, false, false);
    }

    if (n < NCOL) {
        #pragma unroll
        for (int r = 0; r < 8; ++r) {
            int mm = Mbase + r + (hi ? 8 : 0);
            vposed[(size_t)mm * NCOL + n] = c[r];
        }
    }
}

// ---------------- J = Jreg^T @ v_shaped (first 16 joints) ----------------
__global__ void jreg16_kernel(const float* __restrict__ vsh, const float* __restrict__ Jreg,
                              float* __restrict__ Jout) {
    int b = blockIdx.x, t = threadIdx.x;
    if (t >= 48) return;
    int j = t / 3, cc = t % 3;
    const float* vb = vsh + (size_t)b * NCOL;
    float acc = 0.0f;
    for (int v = 0; v < NVV; ++v) acc += Jreg[v * 21 + j] * vb[3 * v + cc];
    Jout[b * 48 + t] = acc;
}

// ---------------- kinematic chain -> relative bone transforms A ----------------
__device__ inline void write_A(float* dst, const float* M /*3x4*/, const float* Jc) {
    #pragma unroll
    for (int r = 0; r < 3; ++r) {
        float trel = M[r*4+3] - (M[r*4+0]*Jc[0] + M[r*4+1]*Jc[1] + M[r*4+2]*Jc[2]);
        dst[r*4+0] = M[r*4+0]; dst[r*4+1] = M[r*4+1]; dst[r*4+2] = M[r*4+2]; dst[r*4+3] = trel;
    }
    dst[12] = 0.0f; dst[13] = 0.0f; dst[14] = 0.0f; dst[15] = 1.0f;
}

__global__ void rigid_kernel(const float* __restrict__ R_all, const float* __restrict__ Jin,
                             float* __restrict__ Aout) {
    int b = blockIdx.x * blockDim.x + threadIdx.x;
    if (b >= NB_) return;
    const int par[16] = {-1, 0, 1, 2, 0, 4, 5, 0, 7, 8, 0, 10, 11, 0, 13, 14};
    const float* R = R_all + (size_t)b * 144;
    const float* J = Jin + (size_t)b * 48;
    float* Ab = Aout + (size_t)b * 256;

    float res0[12], cur[12];
    #pragma unroll
    for (int r = 0; r < 3; ++r) {
        res0[r*4+0] = R[r*3+0]; res0[r*4+1] = R[r*3+1]; res0[r*4+2] = R[r*3+2];
        res0[r*4+3] = J[r];
    }
    write_A(Ab, res0, J);

    for (int i = 1; i < 16; ++i) {
        int p = par[i];
        const float* Ri = R + i * 9;
        const float* Ji = J + i * 3;
        const float* Jp = J + p * 3;
        float Ai[12];
        #pragma unroll
        for (int r = 0; r < 3; ++r) {
            Ai[r*4+0] = Ri[r*3+0]; Ai[r*4+1] = Ri[r*3+1]; Ai[r*4+2] = Ri[r*3+2];
            Ai[r*4+3] = Ji[r] - Jp[r];
        }
        const float* Pm = (p == 0) ? res0 : cur;   // chain resets exactly where parent==0
        float nxt[12];
        #pragma unroll
        for (int r = 0; r < 3; ++r) {
            #pragma unroll
            for (int ccol = 0; ccol < 4; ++ccol) {
                float acc = Pm[r*4+0]*Ai[0*4+ccol] + Pm[r*4+1]*Ai[1*4+ccol] + Pm[r*4+2]*Ai[2*4+ccol];
                if (ccol == 3) acc += Pm[r*4+3];
                nxt[r*4+ccol] = acc;
            }
        }
        #pragma unroll
        for (int e = 0; e < 12; ++e) cur[e] = nxt[e];
        write_A(Ab + i * 16, cur, Ji);
    }
}

// ---------------- skinning: verts = (sum_j w*A_j) @ [v_posed,1] ----------------
__global__ __launch_bounds__(256) void skin_kernel(const float* __restrict__ Aout,
                                                   const float* __restrict__ weights,
                                                   const float* __restrict__ vposed,
                                                   float* __restrict__ verts) {
    int b = blockIdx.y;
    int t = threadIdx.x;
    __shared__ float sA[256];
    sA[t] = Aout[(size_t)b * 256 + t];
    __syncthreads();
    int v = blockIdx.x * 256 + t;
    if (v >= NVV) return;
    float T[12];
    #pragma unroll
    for (int e = 0; e < 12; ++e) T[e] = 0.0f;
    #pragma unroll
    for (int j = 0; j < 16; ++j) {
        float w = weights[v * 16 + j];
        const float* a = sA + j * 16;
        #pragma unroll
        for (int e = 0; e < 12; ++e) T[e] += w * a[e];
    }
    const float* pv = vposed + (size_t)b * NCOL + 3 * v;
    float px = pv[0], py = pv[1], pz = pv[2];
    float* out = verts + (size_t)b * NCOL + 3 * v;
    #pragma unroll
    for (int r = 0; r < 3; ++r)
        out[r] = T[r*4+0]*px + T[r*4+1]*py + T[r*4+2]*pz + T[r*4+3];
}

// ---------------- final joints = Jreg^T @ verts (21 joints) -> d_out ----------------
__global__ void joints_kernel(const float* __restrict__ verts, const float* __restrict__ Jreg,
                              float* __restrict__ out) {
    int b = blockIdx.x, t = threadIdx.x;
    if (t >= 63) return;
    int j = t / 3, cc = t % 3;
    const float* vb = verts + (size_t)b * NCOL;
    float acc = 0.0f;
    for (int v = 0; v < NVV; ++v) acc += Jreg[v * 21 + j] * vb[3 * v + cc];
    out[b * 63 + t] = acc;
}

extern "C" void kernel_launch(void* const* d_in, const int* in_sizes, int n_in,
                              void* d_out, int out_size, void* d_ws, size_t ws_size,
                              hipStream_t stream) {
    if (ws_size < WS_TOTAL) return;
    const float* beta       = (const float*)d_in[0];
    const float* theta      = (const float*)d_in[1];
    const float* quat       = (const float*)d_in[2];
    const float* v_template = (const float*)d_in[3];
    const float* shapedirs  = (const float*)d_in[4];
    const float* posedirs   = (const float*)d_in[5];
    const float* Jreg       = (const float*)d_in[6];
    const float* hands_comp = (const float*)d_in[7];
    const float* hands_mean = (const float*)d_in[8];
    const float* weights    = (const float*)d_in[9];
    float* out = (float*)d_out;

    char* ws = (char*)d_ws;
    float*    R_all  = (float*)   (ws + OFF_RALL);
    _Float16* pf16   = (_Float16*)(ws + OFF_PF16);
    _Float16* beta16 = (_Float16*)(ws + OFF_BETA16);
    _Float16* S16    = (_Float16*)(ws + OFF_S16);
    _Float16* P16    = (_Float16*)(ws + OFF_P16);
    float*    vsh    = (float*)   (ws + OFF_VSH);
    float*    vpo    = (float*)   (ws + OFF_VPO);
    float*    Jws    = (float*)   (ws + OFF_J);
    float*    Amat   = (float*)   (ws + OFF_A);
    float*    verts  = (float*)   (ws + OFF_VERTS);

    conv_S_kernel<<<(NCOLP * KS + 255) / 256, 256, 0, stream>>>(shapedirs, S16);
    conv_P_kernel<<<(NCOLP * KP + 255) / 256, 256, 0, stream>>>(posedirs, P16);
    prep_kernel<<<NB_, 64, 0, stream>>>(theta, quat, beta, hands_comp, hands_mean,
                                        R_all, pf16, beta16);
    gemm_kernel<<<dim3((NTILES_N + 7) / 8, NB_ / 16), 256, 0, stream>>>(beta16, pf16, S16, P16,
                                                                        v_template, vsh, vpo);
    jreg16_kernel<<<NB_, 64, 0, stream>>>(vsh, Jreg, Jws);
    rigid_kernel<<<(NB_ + 63) / 64, 64, 0, stream>>>(R_all, Jws, Amat);
    skin_kernel<<<dim3((NVV + 255) / 256, NB_), 256, 0, stream>>>(Amat, weights, vpo, verts);
    joints_kernel<<<NB_, 64, 0, stream>>>(verts, Jreg, out);
}
